// MPNNReg_80814104641847
// MI455X (gfx1250) — compile-verified, matrix operand order checked by reading, not run on hardware
//
#include <hip/hip_runtime.h>
#include <hip/hip_bf16.h>

// MPNN regression for MI455X (gfx1250, wave32): WMMA bf16 + async global->LDS
// gather (double-buffered) + TDM weight staging.
// N=100000 nodes, E=1.6M edges, H=128, L=4 layers.

typedef __attribute__((ext_vector_type(16))) __bf16 v16bf;
typedef __attribute__((ext_vector_type(8)))  float  v8f;
typedef __attribute__((ext_vector_type(4)))  unsigned int v4u;
typedef __attribute__((ext_vector_type(8)))  int v8i;
typedef __attribute__((ext_vector_type(4)))  int v4i;

#define H    128
#define HPK  16384   // packed elems per 128x128 matrix
#define GPK  49152   // packed elems per 384x128 matrix
#define NLAY 4

__device__ __forceinline__ v8f wmma_bf16(v16bf a, v16bf b, v8f c) {
  // D = A(16x32 bf16) * B(32x16 bf16) + C(16x16 f32)
  return __builtin_amdgcn_wmma_f32_16x16x32_bf16(false, a, false, b, (short)0, c,
                                                 false, false);
}

__device__ __forceinline__ v8f splat8(float x) {
  v8f v;
#pragma unroll
  for (int i = 0; i < 8; ++i) ((float*)&v)[i] = x;
  return v;
}

__device__ __forceinline__ float sigmoid_(float x) {
  return 1.0f / (1.0f + __expf(-x));
}
__device__ __forceinline__ float tanh_(float x) {
  float e = __expf(2.0f * x);
  return (e - 1.0f) / (e + 1.0f);
}

// CDNA5 async copy: GLOBAL_LOAD_ASYNC_TO_LDS_B128 (memory -> LDS, ASYNCcnt).
// Low 32 bits of a generic shared pointer are the raw LDS byte address
// (see ISA 10.2, LDS aperture).
__device__ __forceinline__ void async_copy_b128(void* lds_dst, const void* gsrc) {
  unsigned ldsoff = (unsigned)(unsigned long long)lds_dst;
  asm volatile("global_load_async_to_lds_b128 %0, %1, off"
               :: "v"(ldsoff), "v"(gsrc) : "memory");
}
__device__ __forceinline__ void wait_async0() {
  asm volatile("s_wait_asynccnt 0" ::: "memory");
}

#if __has_builtin(__builtin_amdgcn_tensor_load_to_lds)
#define HAVE_TDM 1
// Tensor Data Mover: 1-D tile (nelem bf16 elements, contiguous) -> LDS.
// D# group0: count=1, lds_addr, global_addr[56:0], type=2 ("image").
// D# group1: data_size=1 (2B), tensor_dim0=tile_dim0=nelem, dim0_stride=nelem.
// This toolchain exposes the 6-arg form:
//   (uint32x4 g0, int32x8 g1, int32x4 g2, int32x4 g3, int32x8 g4, i32 cpol)
__device__ __forceinline__ void tdm_load_1d(void* lds_dst, const void* gsrc,
                                            unsigned nelem) {
  unsigned lds_addr = (unsigned)(unsigned long long)lds_dst;
  unsigned long long ga = (unsigned long long)gsrc;
  v4u g0;
  g0.x = 1u;                                   // count=1 (valid user descriptor)
  g0.y = lds_addr;                             // lds_addr  [63:32]
  g0.z = (unsigned)(ga & 0xffffffffu);         // global_addr lo [95:64]
  g0.w = (unsigned)((ga >> 32) & 0x01ffffffu)  // global_addr hi [120:96]
         | (2u << 30);                         // type=2 [127:126]
  v8i g1;
  g1[0] = (int)(1u << 16);                          // data_size=1 -> 2 bytes
  g1[1] = (int)((nelem & 0xffffu) << 16);           // tensor_dim0 lo [63:48]
  g1[2] = (int)(((nelem >> 16) & 0xffffu) | (1u << 16)); // dim0 hi, tensor_dim1=1
  g1[3] = (int)((nelem & 0xffffu) << 16);           // tile_dim0 [127:112]
  g1[4] = 0;                                        // tile_dim1/2 unused
  g1[5] = (int)nelem;                               // tensor_dim0_stride lo
  g1[6] = 0;
  g1[7] = 0;
  v4i gz4 = {0, 0, 0, 0};
  v8i gz8 = {0, 0, 0, 0, 0, 0, 0, 0};
  __builtin_amdgcn_tensor_load_to_lds(g0, g1, gz4, gz4, gz8, 0);
}
#endif

// Pack a row-major f32 weight matrix W[nOut][128] into bf16 "B-operand" order:
// tile (kt,nt) holds B[k][n] for k in [kt*32,kt*32+32), n in [nt*16,nt*16+16),
// laid out as 32 lanes x 16 contiguous bf16 so a lane issues ONE 32B load.
__global__ void pack_b_kernel(const float* __restrict__ src, __bf16* __restrict__ dst,
                              int NT, int total) {
  int p = blockIdx.x * blockDim.x + threadIdx.x;
  if (p >= total) return;
  int e    = p & 15;
  int lane = (p >> 4) & 31;
  int t    = p >> 9;
  int nt   = t % NT;
  int kt   = t / NT;
  int n = nt * 16 + (lane & 15);
  int k = kt * 32 + (lane >> 4) * 16 + e;
  dst[p] = (__bf16)src[n * H + k];
}

// h = relu(x @ W_in^T + b_in); also produce bf16 shadow for WMMA operands.
__global__ void __launch_bounds__(128)
in_proj_kernel(const float* __restrict__ x, const float* __restrict__ Win,
               const float* __restrict__ bin, float* __restrict__ h,
               __bf16* __restrict__ hbf) {
  __shared__ float lx[64];
  int n = blockIdx.x;
  int f = threadIdx.x;
  if (f < 64) lx[f] = x[(size_t)n * 64 + f];
  __syncthreads();
  float acc = bin[f];
#pragma unroll
  for (int k = 0; k < 64; ++k) acc = fmaf(lx[k], Win[f * 64 + k], acc);
  acc = fmaxf(acc, 0.0f);
  size_t o = (size_t)n * H + f;
  h[o]   = acc;
  hbf[o] = (__bf16)acc;
}

__global__ void zero_aggr_kernel(float* __restrict__ aggr, size_t total) {
  size_t i = (size_t)blockIdx.x * blockDim.x + threadIdx.x;
  if (i < total) aggr[i] = 0.0f;
}

__global__ void zero_sums_kernel(float* __restrict__ s1, float* __restrict__ s2) {
  int i = threadIdx.x;
  if (i < H) { s1[i] = 0.0f; s2[i] = 0.0f; }
}

// Persistent fused gather -> 2-layer MLP (WMMA bf16) -> scatter-add.
// 128 threads = 4 waves. W1+W2 staged in LDS once per block via TDM;
// per-tile gathers are double-buffered async global->LDS copies.
__global__ void __launch_bounds__(128)
message_kernel(const __bf16* __restrict__ hbf,
               const int* __restrict__ src_idx, const int* __restrict__ dst_idx,
               const __bf16* __restrict__ w1p, const __bf16* __restrict__ w2p,
               const float* __restrict__ b1, const float* __restrict__ b2,
               float* __restrict__ aggr, int E, int numTiles) {
  __shared__ __align__(32) __bf16 lw1[HPK];            // 32KB W1 (B-operand order)
  __shared__ __align__(32) __bf16 lw2[HPK];            // 32KB W2
  __shared__ __align__(32) __bf16 lds_x[2][64 * H];    // 2x16KB gathered rows
  __shared__ __align__(32) __bf16 lds_m1[4][16 * H];   // 16KB per-wave relu(m1)
  __shared__ int lds_dst[2][64];

  const int tid    = threadIdx.x;
  const int lane   = tid & 31;
  const int w      = tid >> 5;
  const int laneLo = lane & 15;
  const int laneHi = lane >> 4;

  // ---- stage packed weights into LDS once ----
#if defined(HAVE_TDM)
  if (tid < 32) {               // one wave drives the Tensor Data Mover
    tdm_load_1d(lw1, w1p, HPK);
    tdm_load_1d(lw2, w2p, HPK);
  }
#else
  for (int i = tid; i < HPK / 8; i += 128) {
    async_copy_b128(lw1 + i * 8, w1p + i * 8);
    async_copy_b128(lw2 + i * 8, w2p + i * 8);
  }
#endif

  // per-lane bias values (loop-invariant)
  float bb1[8], bb2[8];
#pragma unroll
  for (int nt = 0; nt < 8; ++nt) {
    bb1[nt] = b1[nt * 16 + laneLo];
    bb2[nt] = b2[nt * 16 + laneLo];
  }

  // async gather of one 64-edge tile into buffer `buf`
  auto prefetch_tile = [&](int t, int buf) {
    int row  = tid >> 1;
    int half = tid & 1;
    int e = t * 64 + row;
    int s = (e < E) ? src_idx[e] : 0;
    const __bf16* g = hbf + (size_t)s * H + half * 64;
    __bf16* l = &lds_x[buf][row * H + half * 64];
#pragma unroll
    for (int i = 0; i < 8; ++i) async_copy_b128(l + i * 8, g + i * 8);
    if (tid < 64) {
      int e2 = t * 64 + tid;
      lds_dst[buf][tid] = (e2 < E) ? dst_idx[e2] : 0;
    }
  };

  int tile = blockIdx.x;
  int buf  = 0;
  if (tile < numTiles) prefetch_tile(tile, 0);
#if defined(HAVE_TDM)
  if (tid < 32) __builtin_amdgcn_s_wait_tensorcnt(0);
#endif

  for (; tile < numTiles; tile += gridDim.x, buf ^= 1) {
    wait_async0();
    __syncthreads();  // gather buf + weights visible to whole block

    int next = tile + gridDim.x;
    if (next < numTiles) prefetch_tile(next, buf ^ 1);  // overlap with compute

    const int edge0 = tile * 64;
    const __bf16* xb = lds_x[buf];

    // ---- GEMM 1: m1 = relu(xj @ W1^T + b1), 16 edges x 128 per wave ----
    v16bf a[4];
#pragma unroll
    for (int kt = 0; kt < 4; ++kt)
      a[kt] = *(const v16bf*)(xb + (w * 16 + laneLo) * H + kt * 32 + laneHi * 16);

    v8f acc[8];
#pragma unroll
    for (int nt = 0; nt < 8; ++nt) acc[nt] = splat8(bb1[nt]);
#pragma unroll
    for (int kt = 0; kt < 4; ++kt) {
      v16bf bt[8];
#pragma unroll
      for (int nt = 0; nt < 8; ++nt)
        bt[nt] = *(const v16bf*)(lw1 + ((kt * 8 + nt) * 512) + lane * 16);
#pragma unroll
      for (int nt = 0; nt < 8; ++nt)
        acc[nt] = wmma_bf16(a[kt], bt[nt], acc[nt]);
    }
    // relu + re-layout through per-wave LDS (D-layout -> row-major 16x128 bf16)
#pragma unroll
    for (int nt = 0; nt < 8; ++nt) {
#pragma unroll
      for (int r = 0; r < 8; ++r) {
        int m = r + laneHi * 8;
        float v = fmaxf(((float*)&acc[nt])[r], 0.0f);
        lds_m1[w][m * H + nt * 16 + laneLo] = (__bf16)v;
      }
    }

    // ---- GEMM 2: m = m1 @ W2^T + b2 ----
    v16bf a2[4];
#pragma unroll
    for (int kt = 0; kt < 4; ++kt)
      a2[kt] = *(const v16bf*)(&lds_m1[w][laneLo * H + kt * 32 + laneHi * 16]);

    v8f acc2[8];
#pragma unroll
    for (int nt = 0; nt < 8; ++nt) acc2[nt] = splat8(bb2[nt]);
#pragma unroll
    for (int kt = 0; kt < 4; ++kt) {
      v16bf bt[8];
#pragma unroll
      for (int nt = 0; nt < 8; ++nt)
        bt[nt] = *(const v16bf*)(lw2 + ((kt * 8 + nt) * 512) + lane * 16);
#pragma unroll
      for (int nt = 0; nt < 8; ++nt)
        acc2[nt] = wmma_bf16(a2[kt], bt[nt], acc2[nt]);
    }

    // ---- scatter-add straight from D layout: lane=col, VGPR=row ----
#pragma unroll
    for (int nt = 0; nt < 8; ++nt) {
      int f = nt * 16 + laneLo;
#pragma unroll
      for (int r = 0; r < 8; ++r) {
        int m = r + laneHi * 8;
        int e = edge0 + w * 16 + m;
        if (e < E) {
          int drow = lds_dst[buf][w * 16 + m];
          atomicAdd(&aggr[(size_t)drow * H + f], ((float*)&acc2[nt])[r]);
        }
      }
    }
    __syncthreads();  // everyone done with buf before it is refilled next round
  }
}

// GRU cell via WMMA (gi = aggr@Wih^T, gh = h@Whh^T), fused gate math,
// writes h_pre in place over aggr, accumulates BN sum/sumsq.
// 256 threads = 8 waves; each wave owns a 16-node tile.
__global__ void __launch_bounds__(256)
gru_kernel(float* __restrict__ aggr, const float* __restrict__ h,
           const __bf16* __restrict__ hbf,
           const __bf16* __restrict__ wihp, const __bf16* __restrict__ whhp,
           const float* __restrict__ bih, const float* __restrict__ bhh,
           float* __restrict__ bnsum, float* __restrict__ bnsumsq, int nTiles) {
  __shared__ __align__(32) __bf16 la[8][16 * H];

  const int tid    = threadIdx.x;
  const int lane   = tid & 31;
  const int w      = tid >> 5;
  const int laneLo = lane & 15;
  const int laneHi = lane >> 4;

  int tile = blockIdx.x * 8 + w;
  const bool active = (tile < nTiles);
  if (!active) tile = nTiles - 1;  // uniform per wave; EXEC stays all-1 for WMMA
  const size_t rowbase = (size_t)tile * 16 * H;

  __builtin_prefetch(wihp + lane * 16, 0, 0);
  __builtin_prefetch(whhp + lane * 16, 0, 0);

  // stage aggr tile as bf16 A operand
  for (int i = lane; i < 16 * H; i += 32)
    la[w][i] = (__bf16)aggr[rowbase + i];

  v16bf aA[4], aH[4];
#pragma unroll
  for (int kt = 0; kt < 4; ++kt) {
    aA[kt] = *(const v16bf*)(&la[w][laneLo * H + kt * 32 + laneHi * 16]);
    aH[kt] = *(const v16bf*)(hbf + rowbase + laneLo * H + kt * 32 + laneHi * 16);
  }

#pragma unroll
  for (int nt = 0; nt < 8; ++nt) {
    int f = nt * 16 + laneLo;
    v8f ir = splat8(bih[f]);
    v8f iz = splat8(bih[H + f]);
    v8f ig = splat8(bih[2 * H + f]);
    v8f hr = splat8(bhh[f]);
    v8f hz = splat8(bhh[H + f]);
    v8f hg = splat8(bhh[2 * H + f]);
#pragma unroll
    for (int kt = 0; kt < 4; ++kt) {
      v16bf bt[6];
      bt[0] = *(const v16bf*)(wihp + ((kt * 24 + nt)      * 512) + lane * 16);
      bt[1] = *(const v16bf*)(wihp + ((kt * 24 + nt + 8)  * 512) + lane * 16);
      bt[2] = *(const v16bf*)(wihp + ((kt * 24 + nt + 16) * 512) + lane * 16);
      bt[3] = *(const v16bf*)(whhp + ((kt * 24 + nt)      * 512) + lane * 16);
      bt[4] = *(const v16bf*)(whhp + ((kt * 24 + nt + 8)  * 512) + lane * 16);
      bt[5] = *(const v16bf*)(whhp + ((kt * 24 + nt + 16) * 512) + lane * 16);
      ir = wmma_bf16(aA[kt], bt[0], ir);
      iz = wmma_bf16(aA[kt], bt[1], iz);
      ig = wmma_bf16(aA[kt], bt[2], ig);
      hr = wmma_bf16(aH[kt], bt[3], hr);
      hz = wmma_bf16(aH[kt], bt[4], hz);
      hg = wmma_bf16(aH[kt], bt[5], hg);
    }
    float s1 = 0.0f, s2 = 0.0f;
#pragma unroll
    for (int r = 0; r < 8; ++r) {
      int m = r + laneHi * 8;
      float rg = sigmoid_(((float*)&ir)[r] + ((float*)&hr)[r]);
      float zg = sigmoid_(((float*)&iz)[r] + ((float*)&hz)[r]);
      float ng = tanh_(((float*)&ig)[r] + rg * ((float*)&hg)[r]);
      float hold = h[rowbase + (size_t)m * H + f];
      float hnew = (1.0f - zg) * ng + zg * hold;
      if (active) aggr[rowbase + (size_t)m * H + f] = hnew;
      s1 += hnew;
      s2 += hnew * hnew;
    }
    if (active) {
      atomicAdd(&bnsum[f], s1);
      atomicAdd(&bnsumsq[f], s2);
    }
  }
}

// BatchNorm (batch stats from fused sums) + residual; refresh bf16 shadow.
// Also zeroes `hpre` (== aggr) in place so the next layer needs no zero pass.
__global__ void bn_kernel(float* __restrict__ hpre, float* __restrict__ h,
                          __bf16* __restrict__ hbf,
                          const float* __restrict__ s1, const float* __restrict__ s2,
                          const float* __restrict__ gamma, const float* __restrict__ beta,
                          float invN, size_t total) {
  size_t i = (size_t)blockIdx.x * blockDim.x + threadIdx.x;
  if (i >= total) return;
  int f = (int)(i & (H - 1));
  float mu  = s1[f] * invN;
  float var = s2[f] * invN - mu * mu;
  float v = (hpre[i] - mu) * rsqrtf(var + 1e-5f);
  hpre[i] = 0.0f;  // ready as `aggr` for the next layer
  float r = gamma[f] * v + beta[f] + h[i];
  h[i]   = r;
  hbf[i] = (__bf16)r;
}

__global__ void out_kernel(const float* __restrict__ h, const float* __restrict__ Wout,
                           const float* __restrict__ bout, float* __restrict__ out,
                           int N) {
  int n = blockIdx.x * blockDim.x + threadIdx.x;
  if (n >= N) return;
  const float4* hr = (const float4*)(h + (size_t)n * H);
  const float4* wr = (const float4*)Wout;
  float acc = 0.0f;
#pragma unroll
  for (int i = 0; i < 32; ++i) {
    float4 a = hr[i], b = wr[i];
    acc = fmaf(a.x, b.x, fmaf(a.y, b.y, fmaf(a.z, b.z, fmaf(a.w, b.w, acc))));
  }
  out[n] = acc + bout[0];
}

extern "C" void kernel_launch(void* const* d_in, const int* in_sizes, int n_in,
                              void* d_out, int out_size, void* d_ws, size_t ws_size,
                              hipStream_t stream) {
  const float* x     = (const float*)d_in[0];
  const int*   ei    = (const int*)d_in[1];
  const float* Win   = (const float*)d_in[2];
  const float* bin   = (const float*)d_in[3];
  const float* mW1   = (const float*)d_in[4];
  const float* mb1   = (const float*)d_in[5];
  const float* mW2   = (const float*)d_in[6];
  const float* mb2   = (const float*)d_in[7];
  const float* gWih  = (const float*)d_in[8];
  const float* gbih  = (const float*)d_in[9];
  const float* gWhh  = (const float*)d_in[10];
  const float* gbhh  = (const float*)d_in[11];
  const float* gamma = (const float*)d_in[12];
  const float* beta  = (const float*)d_in[13];
  const float* Wout  = (const float*)d_in[14];
  const float* bout  = (const float*)d_in[15];

  const int N = in_sizes[0] / 64;
  const int E = in_sizes[1] / 2;

  char* ws = (char*)d_ws;
  size_t off = 0;
  auto alloc = [&](size_t bytes) -> char* {
    char* p = ws + off;
    off += (bytes + 255) & ~(size_t)255;
    return p;
  };
  float*  h    = (float*)alloc((size_t)N * H * 4);
  __bf16* hbf  = (__bf16*)alloc((size_t)N * H * 2);
  float*  aggr = (float*)alloc((size_t)N * H * 4);
  __bf16* w1p  = (__bf16*)alloc((size_t)NLAY * HPK * 2);
  __bf16* w2p  = (__bf16*)alloc((size_t)NLAY * HPK * 2);
  __bf16* wihp = (__bf16*)alloc((size_t)NLAY * GPK * 2);
  __bf16* whhp = (__bf16*)alloc((size_t)NLAY * GPK * 2);
  float*  bns1 = (float*)alloc(H * 4);
  float*  bns2 = (float*)alloc(H * 4);

  // Pre-pack all weights into WMMA B-operand bf16 layout (tiny, once per call).
  for (int l = 0; l < NLAY; ++l) {
    pack_b_kernel<<<HPK / 256, 256, 0, stream>>>(mW1 + (size_t)l * H * H,
                                                 w1p + (size_t)l * HPK, 8, HPK);
    pack_b_kernel<<<HPK / 256, 256, 0, stream>>>(mW2 + (size_t)l * H * H,
                                                 w2p + (size_t)l * HPK, 8, HPK);
    pack_b_kernel<<<GPK / 256, 256, 0, stream>>>(gWih + (size_t)l * 3 * H * H,
                                                 wihp + (size_t)l * GPK, 24, GPK);
    pack_b_kernel<<<GPK / 256, 256, 0, stream>>>(gWhh + (size_t)l * 3 * H * H,
                                                 whhp + (size_t)l * GPK, 24, GPK);
  }

  in_proj_kernel<<<N, 128, 0, stream>>>(x, Win, bin, h, hbf);

  const size_t total   = (size_t)N * H;
  const int nTiles     = (N + 15) / 16;
  const int msgTiles   = (E + 63) / 64;
  const int msgBlocks  = msgTiles < 1536 ? msgTiles : 1536;  // persistent blocks
  const int gruBlocks  = (nTiles + 7) / 8;
  const int elemBlocks = (int)((total + 255) / 256);
  const float invN     = 1.0f / (float)N;

  zero_aggr_kernel<<<elemBlocks, 256, 0, stream>>>(aggr, total);  // layer 0 only
  for (int l = 0; l < NLAY; ++l) {
    message_kernel<<<msgBlocks, 128, 0, stream>>>(
        hbf, ei, ei + E, w1p + (size_t)l * HPK, w2p + (size_t)l * HPK,
        mb1 + l * H, mb2 + l * H, aggr, E, msgTiles);
    zero_sums_kernel<<<1, 128, 0, stream>>>(bns1, bns2);
    gru_kernel<<<gruBlocks, 256, 0, stream>>>(
        aggr, h, hbf, wihp + (size_t)l * GPK, whhp + (size_t)l * GPK,
        gbih + l * 3 * H, gbhh + l * 3 * H, bns1, bns2, nTiles);
    bn_kernel<<<elemBlocks, 256, 0, stream>>>(aggr, h, hbf, bns1, bns2,
                                              gamma + l * H, beta + l * H,
                                              invN, total);
  }
  out_kernel<<<(N + 255) / 256, 256, 0, stream>>>(h, Wout, bout, (float*)d_out, N);
}
